// GlobalAttention_23476291239961
// MI455X (gfx1250) — matrix operand order repaired and checked by reading, not compiled, not run on hardware
//
#include <hip/hip_runtime.h>
#include <hip/hip_bf16.h>

// ---------------------------------------------------------------------------
// GlobalAttention (AlphaFold-style) for MI455X / gfx1250, wave32 + WMMA bf16.
// B=512, N=2048, QD=MD=64, H=8, KD=VD=8, OD=64.
// ---------------------------------------------------------------------------

#define GB_ 512
#define GN_ 2048
#define GD_ 64

typedef __attribute__((ext_vector_type(16))) __bf16 v16bf;
typedef __attribute__((ext_vector_type(8)))  float  v8f;

union U16bf {
    unsigned      u[8];
    uint4         q[2];
    v16bf         v;
};

// round-to-nearest-even f32 -> bf16 pair pack
__device__ inline unsigned pk_bf16(float a, float b) {
    unsigned ua = __builtin_bit_cast(unsigned, a);
    unsigned ub = __builtin_bit_cast(unsigned, b);
    ua += 0x7FFFu + ((ua >> 16) & 1u);
    ub += 0x7FFFu + ((ub >> 16) & 1u);
    return (ua >> 16) | (ub & 0xFFFF0000u);
}
__device__ inline unsigned short bf16b(float a) {
    unsigned ua = __builtin_bit_cast(unsigned, a);
    ua += 0x7FFFu + ((ua >> 16) & 1u);
    return (unsigned short)(ua >> 16);
}

// A operand (16x32 bf16) from a global f32 row. Lane holds row = lane&15.
// kb = Kofs + khalf*8 ; spans [kb,kb+8) and [kb+16,kb+24) per ISA A layout.
__device__ inline v16bf ldA_g(const float* __restrict__ rowp, int kb) {
    float4 x0 = *(const float4*)(rowp + kb);
    float4 x1 = *(const float4*)(rowp + kb + 4);
    float4 y0 = *(const float4*)(rowp + kb + 16);
    float4 y1 = *(const float4*)(rowp + kb + 20);
    U16bf r;
    r.u[0] = pk_bf16(x0.x, x0.y); r.u[1] = pk_bf16(x0.z, x0.w);
    r.u[2] = pk_bf16(x1.x, x1.y); r.u[3] = pk_bf16(x1.z, x1.w);
    r.u[4] = pk_bf16(y0.x, y0.y); r.u[5] = pk_bf16(y0.z, y0.w);
    r.u[6] = pk_bf16(y1.x, y1.y); r.u[7] = pk_bf16(y1.z, y1.w);
    return r.v;
}

// B operand (32x16 bf16) from LDS (weights stored transposed: [col][k]).
// ks = Kofs + khalf*8 ; spans [ks,ks+8) and [ks+16,ks+24) per ISA layout.
__device__ inline v16bf ldB_s(const unsigned short* p, int ks) {
    U16bf r;
    r.q[0] = *(const uint4*)(p + ks);
    r.q[1] = *(const uint4*)(p + ks + 16);
    return r.v;
}

// A operand from a bf16 LDS staging row (row stride must be 16B aligned).
__device__ inline v16bf ldA_s(const unsigned short* rowp, int kb) {
    U16bf r;
    r.q[0] = *(const uint4*)(rowp + kb);
    r.q[1] = *(const uint4*)(rowp + kb + 16);
    return r.v;
}

__device__ inline v8f wmma_bf16(v16bf a, v16bf b, v8f c) {
    return __builtin_amdgcn_wmma_f32_16x16x32_bf16(false, a, false, b,
                                                   (short)0, c, false, false);
}

// ---------------------------------------------------------------------------
// Kernel 1: masked partial sums of q_data over N-segments (4096 blocks).
// part[(b*8+seg)*65 + c] = sum_n mask*q_data ; [.. + 64] = sum_n mask
// ---------------------------------------------------------------------------
__global__ __launch_bounds__(256) void ga_qsum_kernel(
    const float* __restrict__ q_data, const float* __restrict__ q_mask,
    float* __restrict__ part)
{
    __shared__ float sSum[4][64];
    __shared__ float sM[4];
    const int tid = threadIdx.x;
    const int b = blockIdx.x >> 3, seg = blockIdx.x & 7;
    const int c = tid & 63, r = tid >> 6;
    const int nbeg = seg * 256;
    float acc = 0.f, macc = 0.f;
    for (int n = nbeg + r; n < nbeg + 256; n += 4) {
        float mk = q_mask[(size_t)b * GN_ + n];
        acc += mk * q_data[((size_t)b * GN_ + n) * GD_ + c];
        macc += mk;
    }
    sSum[r][c] = acc;
    if (c == 0) sM[r] = macc;
    __syncthreads();
    if (r == 0) {
        float tot = sSum[0][c] + sSum[1][c] + sSum[2][c] + sSum[3][c];
        part[(size_t)(b * 8 + seg) * 65 + c] = tot;
        if (c == 0)
            part[(size_t)(b * 8 + seg) * 65 + 64] = sM[0] + sM[1] + sM[2] + sM[3];
    }
}

// ---------------------------------------------------------------------------
// Kernel 2: q_avg -> q = (q_avg @ query_w) * KD^-0.5   (512 blocks x 64 thr)
// ---------------------------------------------------------------------------
__global__ __launch_bounds__(64) void ga_qproj_kernel(
    const float* __restrict__ part, const float* __restrict__ query_w,
    float* __restrict__ qh)
{
    __shared__ float sAvg[64];
    const int tid = threadIdx.x;
    const int b = blockIdx.x;
    float tot = 0.f, den = 1e-10f;
    for (int s = 0; s < 8; ++s) {
        tot += part[(size_t)(b * 8 + s) * 65 + tid];
        den += part[(size_t)(b * 8 + s) * 65 + 64];
    }
    sAvg[tid] = tot / den;
    __syncthreads();
    float s = 0.f;
    for (int a = 0; a < 64; ++a) s += sAvg[a] * query_w[a * 64 + tid];
    qh[(size_t)b * 64 + tid] = s * 0.35355339059327373f;  // 8^-0.5
}

// ---------------------------------------------------------------------------
// Kernel 3: per-batch attention. One block/batch, 8 waves. Each wave:
//  - WMMA 16-row m_data tile against packed [64,16] (key_w | value_w) weights
//  - stage 16x16 kv tile in LDS, flash-style online softmax per head
//  - cross-wave combine -> weighted_avg[b,8,8]
// ---------------------------------------------------------------------------
__global__ __launch_bounds__(256) void ga_attn_kernel(
    const float* __restrict__ m_data, const float* __restrict__ q_mask,
    const float* __restrict__ key_w, const float* __restrict__ value_w,
    const float* __restrict__ qh, float* __restrict__ wavg)
{
    __shared__ __align__(16) unsigned short sWkvT[16][64];  // [col][k]
    __shared__ float sQ[64];
    __shared__ __align__(16) float sStage[8][16][17];       // per-wave kv tile
    __shared__ float sRed[8][8][10];                        // [wave][h][m,s,acc0..7]

    const int tid = threadIdx.x;
    const int b = blockIdx.x;
    const int wave = tid >> 5, lane = tid & 31;
    const int khalf = lane >> 4, col = lane & 15;

    for (int i = tid; i < 1024; i += 256) {
        int kc = i >> 4, cc = i & 15;
        float w = (cc < 8) ? key_w[kc * 8 + cc] : value_w[kc * 8 + (cc - 8)];
        sWkvT[cc][kc] = bf16b(w);
    }
    if (tid < 64) sQ[tid] = qh[(size_t)b * 64 + tid];
    __syncthreads();

    const v16bf Bk0 = ldB_s(&sWkvT[col][0], khalf * 8);
    const v16bf Bk1 = ldB_s(&sWkvT[col][0], 32 + khalf * 8);

    const int h = lane >> 2;            // head owned by this lane
    const int vp = (lane & 3) * 2;      // v-pair owned by this lane
    float mh = -INFINITY, sh = 0.f, a0 = 0.f, a1 = 0.f;

    for (int t = wave; t < GN_ / 16; t += 8) {
        const int n0 = t * 16;
        const float* rowp = m_data + ((size_t)b * GN_ + n0 + col) * GD_;
        v16bf A0 = ldA_g(rowp, khalf * 8);
        v16bf A1 = ldA_g(rowp, 32 + khalf * 8);
        v8f cc = {};
        cc = wmma_bf16(A0, Bk0, cc);
        cc = wmma_bf16(A1, Bk1, cc);
#pragma unroll
        for (int j = 0; j < 8; ++j)
            sStage[wave][khalf * 8 + j][col] = cc[j];
        asm volatile("s_wait_dscnt 0" ::: "memory");

        for (int nn = 0; nn < 16; ++nn) {
            float mb = 32768.0f * (q_mask[(size_t)b * GN_ + n0 + nn] - 1.0f);
            float l = mb;
#pragma unroll
            for (int c8 = 0; c8 < 8; ++c8)
                l += sQ[h * 8 + c8] * sStage[wave][nn][c8];
            float nm = fmaxf(mh, l);
            float corr = __expf(mh - nm);     // exp(-inf)=0 on first tile
            float e = __expf(l - nm);
            sh = sh * corr + e;
            a0 = a0 * corr + e * sStage[wave][nn][8 + vp];
            a1 = a1 * corr + e * sStage[wave][nn][9 + vp];
            mh = nm;
        }
    }

    if ((lane & 3) == 0) { sRed[wave][h][0] = mh; sRed[wave][h][1] = sh; }
    sRed[wave][h][2 + vp] = a0;
    sRed[wave][h][3 + vp] = a1;
    __syncthreads();

    if (tid < 64) {
        const int hh = tid >> 3, vv = tid & 7;
        float M = -INFINITY;
        for (int w = 0; w < 8; ++w) M = fmaxf(M, sRed[w][hh][0]);
        float S = 0.f, A = 0.f;
        for (int w = 0; w < 8; ++w) {
            float cw = __expf(sRed[w][hh][0] - M);
            S += sRed[w][hh][1] * cw;
            A += sRed[w][hh][2 + vv] * cw;
        }
        wavg[(size_t)b * 64 + tid] = A / S;
    }
}

// ---------------------------------------------------------------------------
// Kernel 4: gate + output projection. 8192 blocks x 8 waves; each wave owns a
// 16-row tile: WMMA(q_data, gating_w) -> sigmoid gate * weighted_avg ->
// WMMA(gated, output_w) + bias -> out.
// ---------------------------------------------------------------------------
__global__ __launch_bounds__(256) void ga_gateout_kernel(
    const float* __restrict__ q_data, const float* __restrict__ gating_w,
    const float* __restrict__ gating_b, const float* __restrict__ output_w,
    const float* __restrict__ output_b, const float* __restrict__ wavg,
    float* __restrict__ out)
{
    __shared__ __align__(16) unsigned short sW1T[64][64];      // [hv][c]
    __shared__ __align__(16) unsigned short sW2T[64][64];      // [o][hv]
    __shared__ __align__(16) unsigned short sStage[8][16][72]; // gated bf16
    __shared__ float sGb[64], sOb[64], sWv[64];

    const int tid = threadIdx.x;
    const int b = blockIdx.x >> 4;
    const int nblk = (blockIdx.x & 15) * 128;
    const int wave = tid >> 5, lane = tid & 31;
    const int khalf = lane >> 4, col = lane & 15;

    for (int i = tid; i < 4096; i += 256) {
        int c = i >> 6, hv = i & 63;
        sW1T[hv][c] = bf16b(gating_w[c * 64 + hv]);
    }
    for (int i = tid; i < 4096; i += 256) {
        int hv = i >> 6, o = i & 63;
        sW2T[o][hv] = bf16b(output_w[hv * 64 + o]);
    }
    if (tid < 64) {
        sGb[tid] = gating_b[tid];
        sOb[tid] = output_b[tid];
        sWv[tid] = wavg[(size_t)b * 64 + tid];
    }
    __syncthreads();

    const int n0 = nblk + wave * 16;
    const float* rowp = q_data + ((size_t)b * GN_ + n0 + col) * GD_;
    const v16bf A0 = ldA_g(rowp, khalf * 8);
    const v16bf A1 = ldA_g(rowp, 32 + khalf * 8);

#pragma unroll
    for (int nt = 0; nt < 4; ++nt) {
        const int hv = nt * 16 + col;
        v16bf B0 = ldB_s(&sW1T[hv][0], khalf * 8);
        v16bf B1 = ldB_s(&sW1T[hv][0], 32 + khalf * 8);
        v8f cc = {};
        cc = wmma_bf16(A0, B0, cc);
        cc = wmma_bf16(A1, B1, cc);
        const float wv = sWv[hv], gb = sGb[hv];
#pragma unroll
        for (int j = 0; j < 8; ++j) {
            float g = wv / (1.0f + __expf(-(cc[j] + gb)));
            sStage[wave][khalf * 8 + j][hv] = bf16b(g);
        }
    }
    asm volatile("s_wait_dscnt 0" ::: "memory");

    const v16bf G0 = ldA_s(&sStage[wave][col][0], khalf * 8);
    const v16bf G1 = ldA_s(&sStage[wave][col][0], 32 + khalf * 8);

    float* obase = out + ((size_t)b * GN_ + n0) * GD_;
#pragma unroll
    for (int ot = 0; ot < 4; ++ot) {
        const int o = ot * 16 + col;
        v16bf B0 = ldB_s(&sW2T[o][0], khalf * 8);
        v16bf B1 = ldB_s(&sW2T[o][0], 32 + khalf * 8);
        v8f cc = {};
        cc = wmma_bf16(G0, B0, cc);
        cc = wmma_bf16(G1, B1, cc);
        const float ob = sOb[o];
#pragma unroll
        for (int j = 0; j < 8; ++j)
            obase[(size_t)(khalf * 8 + j) * GD_ + o] = cc[j] + ob;
    }
}

// ---------------------------------------------------------------------------
extern "C" void kernel_launch(void* const* d_in, const int* in_sizes, int n_in,
                              void* d_out, int out_size, void* d_ws, size_t ws_size,
                              hipStream_t stream) {
    const float* q_data   = (const float*)d_in[0];
    const float* m_data   = (const float*)d_in[1];
    const float* q_mask   = (const float*)d_in[2];
    /* d_in[3] = bias (unused by reference) */
    const float* query_w  = (const float*)d_in[4];
    const float* key_w    = (const float*)d_in[5];
    const float* value_w  = (const float*)d_in[6];
    const float* gating_w = (const float*)d_in[7];
    const float* gating_b = (const float*)d_in[8];
    const float* output_w = (const float*)d_in[9];
    const float* output_b = (const float*)d_in[10];
    float* out = (float*)d_out;

    float* ws   = (float*)d_ws;
    float* part = ws;                          // 512*8*65 floats
    float* qh   = ws + (size_t)GB_ * 8 * 65;   // 512*64 floats
    float* wavg = qh + (size_t)GB_ * 64;       // 512*64 floats

    ga_qsum_kernel<<<GB_ * 8, 256, 0, stream>>>(q_data, q_mask, part);
    ga_qproj_kernel<<<GB_, 64, 0, stream>>>(part, query_w, qh);
    ga_attn_kernel<<<GB_, 256, 0, stream>>>(m_data, q_mask, key_w, value_w, qh, wavg);
    ga_gateout_kernel<<<GB_ * 16, 256, 0, stream>>>(q_data, gating_w, gating_b,
                                                    output_w, output_b, wavg, out);
}
